// AudioVQMix_41592463294644
// MI455X (gfx1250) — matrix operand order repaired and checked by reading, not compiled, not run on hardware
//
#include <hip/hip_runtime.h>
#include <hip/hip_bf16.h>
#include <math.h>

typedef __attribute__((ext_vector_type(16))) __bf16 v16bf;
typedef __attribute__((ext_vector_type(8)))  float  v8f;

// ---------------------------------------------------------------------------
// Prep kernels
// ---------------------------------------------------------------------------

__global__ __launch_bounds__(256)
void avq_costab_kernel(float* __restrict__ costab) {
    int t = blockIdx.x * blockDim.x + threadIdx.x;
    if (t < 80 * 38) {
        int n = t / 38, j = t % 38;                       // bin k = j + 2
        costab[t] = cosf(2.0f * 3.14159265358979323846f * (float)((j + 2) * n) / 80.0f);
    }
}

__global__ __launch_bounds__(256)
void avq_convert_pad_bf16_kernel(const float* __restrict__ src, __bf16* __restrict__ dst,
                                 long rows, int dsrc, int dpad) {
    long total = rows * (long)dpad;
    for (long t = blockIdx.x * (long)blockDim.x + threadIdx.x; t < total;
         t += (long)gridDim.x * blockDim.x) {
        long r = t / dpad;
        int  c = (int)(t - r * (long)dpad);
        float v = (c < dsrc) ? src[r * (long)dsrc + c] : 0.0f;
        dst[t] = (__bf16)v;
    }
}

__global__ __launch_bounds__(256)
void avq_row_norm2_kernel(const float* __restrict__ src, float* __restrict__ out,
                          long rows, int d) {
    long r = blockIdx.x * (long)blockDim.x + threadIdx.x;
    if (r >= rows) return;
    const float* p = src + r * (long)d;
    float s = 0.0f;
    for (int i = 0; i < d; ++i) s += p[i] * p[i];
    out[r] = s;
}

// ---------------------------------------------------------------------------
// MFCC transform: l2norm -> real-DFT bins 2..39 (cos matmul) -> l2norm -> bf16
// ---------------------------------------------------------------------------

__global__ __launch_bounds__(256)
void avq_mfcc_transform_kernel(const float* __restrict__ X,
                               const float* __restrict__ costab,   // [80][38]
                               __bf16* __restrict__ xm,            // [NF][64]
                               float* __restrict__ xnorm_m, long nframes) {
    long f = blockIdx.x * (long)blockDim.x + threadIdx.x;
    if (f >= nframes) return;
    const float* xr = X + f * 80;
    float v[80];
    float ss = 0.0f;
    #pragma unroll
    for (int i = 0; i < 80; ++i) { v[i] = xr[i]; ss += v[i] * v[i]; }
    float inv = 1.0f / fmaxf(sqrtf(ss), 1e-12f);
    float spec[38];
    #pragma unroll
    for (int j = 0; j < 38; ++j) spec[j] = 0.0f;
    for (int i = 0; i < 80; ++i) {
        float xi = v[i] * inv;
        const float* cr = costab + i * 38;
        #pragma unroll
        for (int j = 0; j < 38; ++j) spec[j] += xi * cr[j];
    }
    float s2 = 0.0f;
    #pragma unroll
    for (int j = 0; j < 38; ++j) s2 += spec[j] * spec[j];
    float inv2 = 1.0f / fmaxf(sqrtf(s2), 1e-12f);
    __bf16* o = xm + f * 64;
    #pragma unroll
    for (int j = 0; j < 38; ++j) o[j] = (__bf16)(spec[j] * inv2);
    #pragma unroll
    for (int j = 38; j < 64; ++j) o[j] = (__bf16)0.0f;
    xnorm_m[f] = s2 * inv2 * inv2;   // ~1.0 after normalization
}

// ---------------------------------------------------------------------------
// WMMA VQ kernel, M-blocked: MT*16 frames/block, 8 waves, each wave owns
// KCODES/8 codebook columns. Every B fragment loaded once is reused against
// MT A fragments (MT independent WMMA chains -> latency hiding, 4x less L2
// codebook traffic).  score s = x.c via v_wmma_f32_16x16x32_bf16;
// argmin_k (||c||^2 - 2s); d_min = ||x||^2 + key.
// ---------------------------------------------------------------------------

template <int KCODES, int DP, int MT>
__global__ __launch_bounds__(256)
void avq_vq_wmma_kernel(const __bf16* __restrict__ xbf,   // [NF][DP]
                        const float*  __restrict__ xnorm, // [NF]
                        const __bf16* __restrict__ cbbf,  // [KCODES][DP]
                        const float*  __restrict__ cn,    // [KCODES]
                        float* __restrict__ out_enc,      // interleaved (2 per frame)
                        int idx_offset,                   // 0 (ch0) or K_wave (ch1)
                        int* __restrict__ counts,
                        float* __restrict__ dsum) {
    constexpr int KB = DP / 32;              // WMMA K-blocks
    constexpr int NT = KCODES / (8 * 16);    // 16-wide column tiles per wave
    const int tid  = threadIdx.x;
    const int lane = tid & 31;
    const int wave = tid >> 5;
    const int hi   = lane >> 4;              // lane half (wave32)
    const int l15  = lane & 15;
    const long tile_base = (long)blockIdx.x * (16 * MT);

    // --- A fragments: MT tiles of 16x32 bf16 per K-block (ISA 7.12.2) ------
    v16bf a[MT][KB];
    #pragma unroll
    for (int mt = 0; mt < MT; ++mt) {
        const __bf16* xrow = xbf + (tile_base + mt * 16 + l15) * DP;
        #pragma unroll
        for (int kb = 0; kb < KB; ++kb) {
            #pragma unroll
            for (int r = 0; r < 8; ++r) {
                int k0 = kb * 32 + hi * 8 + (r < 4 ? 2 * r : 16 + 2 * (r - 4));
                a[mt][kb][2 * r]     = xrow[k0];
                a[mt][kb][2 * r + 1] = xrow[k0 + 1];
            }
        }
    }

    float bestk[MT][8];
    int   bestn[MT][8];
    #pragma unroll
    for (int mt = 0; mt < MT; ++mt)
        #pragma unroll
        for (int r = 0; r < 8; ++r) { bestk[mt][r] = 3.4e38f; bestn[mt][r] = 0; }

    for (int t = 0; t < NT; ++t) {
        int n = (wave * NT + t) * 16 + l15;              // codebook column
        const __bf16* cbrow = cbbf + (long)n * DP;
        if (t + 1 < NT)                                  // global_prefetch_b8
            __builtin_prefetch(cbbf + (long)(n + 16) * DP, 0, 0);

        v8f acc[MT];
        #pragma unroll
        for (int mt = 0; mt < MT; ++mt)
            acc[mt] = (v8f){0.f, 0.f, 0.f, 0.f, 0.f, 0.f, 0.f, 0.f};

        #pragma unroll
        for (int kb = 0; kb < KB; ++kb) {
            v16bf b;                                     // B 32x16: lanes0-15 K=0..7,16..23
            #pragma unroll
            for (int r = 0; r < 8; ++r) {
                int k0 = kb * 32 + hi * 8 + (r < 4 ? 2 * r : 16 + 2 * (r - 4));
                b[2 * r]     = cbrow[k0];
                b[2 * r + 1] = cbrow[k0 + 1];
            }
            #pragma unroll
            for (int mt = 0; mt < MT; ++mt)              // reuse B across MT chains
                acc[mt] = __builtin_amdgcn_wmma_f32_16x16x32_bf16(
                              false, a[mt][kb], false, b, (short)0, acc[mt], false, false);
        }

        float cnn = cn[n];
        #pragma unroll
        for (int mt = 0; mt < MT; ++mt) {
            #pragma unroll
            for (int r = 0; r < 8; ++r) {                // rows m = hi*8 + r
                float key = cnn - 2.0f * acc[mt][r];
                if (key < bestk[mt][r]) { bestk[mt][r] = key; bestn[mt][r] = n; }
            }
        }
    }

    // reduce over the 16 lanes of each half (xor masks < 16 keep halves intact)
    #pragma unroll
    for (int off = 8; off >= 1; off >>= 1) {
        #pragma unroll
        for (int mt = 0; mt < MT; ++mt) {
            #pragma unroll
            for (int r = 0; r < 8; ++r) {
                float ok = __shfl_xor(bestk[mt][r], off, 32);
                int   on = __shfl_xor(bestn[mt][r], off, 32);
                if (ok < bestk[mt][r]) { bestk[mt][r] = ok; bestn[mt][r] = on; }
            }
        }
    }

    __shared__ float sdist[8][MT * 16];
    __shared__ int   sidx[8][MT * 16];
    if (l15 == 0) {
        #pragma unroll
        for (int mt = 0; mt < MT; ++mt)
            #pragma unroll
            for (int r = 0; r < 8; ++r) {
                sdist[wave][mt * 16 + hi * 8 + r] = bestk[mt][r];
                sidx [wave][mt * 16 + hi * 8 + r] = bestn[mt][r];
            }
    }
    __syncthreads();

    if (tid < MT * 16) {
        float bk = sdist[0][tid];
        int   bn = sidx[0][tid];
        #pragma unroll
        for (int w = 1; w < 8; ++w)
            if (sdist[w][tid] < bk) { bk = sdist[w][tid]; bn = sidx[w][tid]; }
        long f = tile_base + tid;
        float d = xnorm[f] + bk;                         // ||x-c||^2
        if (d < 0.0f) d = 0.0f;
        out_enc[2 * f + (idx_offset ? 1 : 0)] = (float)(bn + idx_offset);
        atomicAdd(&counts[bn], 1);
        atomicAdd(dsum, d);
    }
}

// ---------------------------------------------------------------------------
// Finalize: perplexity from histograms, loss = 1.25 * sum(d_min) / (BN*D)
// ---------------------------------------------------------------------------

__global__ __launch_bounds__(256)
void avq_finalize_kernel(const int* __restrict__ counts_w, const int* __restrict__ counts_m,
                         const float* __restrict__ dsum_w, const float* __restrict__ dsum_m,
                         float* __restrict__ out_scalars, float inv_total) {
    __shared__ float red[256];
    int tid = threadIdx.x;

    float e = 0.0f;
    for (int k = tid; k < 1024; k += 256) {
        float p = (float)counts_w[k] * inv_total;
        e -= p * logf(p + 1e-10f);
    }
    red[tid] = e;
    __syncthreads();
    for (int s = 128; s > 0; s >>= 1) {
        if (tid < s) red[tid] += red[tid + s];
        __syncthreads();
    }
    if (tid == 0) {
        out_scalars[0] = expf(red[0]);                              // wave_ppl
        out_scalars[1] = 1.25f * dsum_w[0] * inv_total / 80.0f;     // wave_loss
    }
    __syncthreads();

    float e2 = 0.0f;
    if (tid < 256) {
        float p = (float)counts_m[tid] * inv_total;
        e2 = -p * logf(p + 1e-10f);
    }
    red[tid] = e2;
    __syncthreads();
    for (int s = 128; s > 0; s >>= 1) {
        if (tid < s) red[tid] += red[tid + s];
        __syncthreads();
    }
    if (tid == 0) {
        out_scalars[2] = expf(red[0]);                              // mfcc_ppl
        out_scalars[3] = 1.25f * dsum_m[0] * inv_total / 38.0f;     // mfcc_loss
    }
}

// ---------------------------------------------------------------------------
// Host launcher
// ---------------------------------------------------------------------------

extern "C" void kernel_launch(void* const* d_in, const int* in_sizes, int n_in,
                              void* d_out, int out_size, void* d_ws, size_t ws_size,
                              hipStream_t stream) {
    (void)n_in; (void)out_size; (void)ws_size;
    const float* X   = (const float*)d_in[0];   // (16,4096,80)
    const float* CBW = (const float*)d_in[1];   // (1024,80)
    const float* CBM = (const float*)d_in[2];   // (256,38)
    float* out = (float*)d_out;

    const long NF = (long)in_sizes[0] / 80;     // 65536 frames
    const int  KW = in_sizes[1] / 80;           // 1024
    const int  KM = in_sizes[2] / 38;           // 256

    // workspace layout (bytes, 256B-aligned chunks)
    char* ws = (char*)d_ws;
    size_t off = 0;
    auto alloc = [&](size_t bytes) { size_t o = off; off = (off + bytes + 255) & ~(size_t)255; return o; };
    __bf16* xw_bf  = (__bf16*)(ws + alloc(NF * 96 * 2));
    float*  xn_w   = (float*) (ws + alloc(NF * 4));
    __bf16* cbw_bf = (__bf16*)(ws + alloc((size_t)KW * 96 * 2));
    float*  cn_w   = (float*) (ws + alloc((size_t)KW * 4));
    __bf16* xm_bf  = (__bf16*)(ws + alloc(NF * 64 * 2));
    float*  xn_m   = (float*) (ws + alloc(NF * 4));
    __bf16* cbm_bf = (__bf16*)(ws + alloc((size_t)KM * 64 * 2));
    float*  cn_m   = (float*) (ws + alloc((size_t)KM * 4));
    size_t zero_off = off;
    int*    cnt_w  = (int*)   (ws + alloc((size_t)KW * 4));
    int*    cnt_m  = (int*)   (ws + alloc((size_t)KM * 4));
    float*  ds_w   = (float*) (ws + alloc(4));
    float*  ds_m   = (float*) (ws + alloc(4));
    size_t zero_len = off - zero_off;
    float*  costab = (float*) (ws + alloc(80 * 38 * 4));

    hipMemsetAsync(ws + zero_off, 0, zero_len, stream);

    // prep
    avq_costab_kernel<<<(80 * 38 + 255) / 256, 256, 0, stream>>>(costab);
    avq_convert_pad_bf16_kernel<<<4096, 256, 0, stream>>>(X, xw_bf, NF, 80, 96);
    avq_row_norm2_kernel<<<(int)((NF + 255) / 256), 256, 0, stream>>>(X, xn_w, NF, 80);
    avq_convert_pad_bf16_kernel<<<512, 256, 0, stream>>>(CBW, cbw_bf, KW, 80, 96);
    avq_row_norm2_kernel<<<(KW + 255) / 256, 256, 0, stream>>>(CBW, cn_w, KW, 80);
    avq_convert_pad_bf16_kernel<<<64, 256, 0, stream>>>(CBM, cbm_bf, KM, 38, 64);
    avq_row_norm2_kernel<<<(KM + 255) / 256, 256, 0, stream>>>(CBM, cn_m, KM, 38);

    // mfcc transform
    avq_mfcc_transform_kernel<<<(int)((NF + 255) / 256), 256, 0, stream>>>(
        X, costab, xm_bf, xn_m, NF);

    // VQ (WMMA), 64 frames per block (MT=4)
    const int blocks = (int)(NF / 64);
    avq_vq_wmma_kernel<1024, 96, 4><<<blocks, 256, 0, stream>>>(
        xw_bf, xn_w, cbw_bf, cn_w, out, /*idx_offset=*/0, cnt_w, ds_w);
    avq_vq_wmma_kernel<256, 64, 4><<<blocks, 256, 0, stream>>>(
        xm_bf, xn_m, cbm_bf, cn_m, out, /*idx_offset=*/KW, cnt_m, ds_m);

    // scalars after the 2*NF interleaved encodings
    avq_finalize_kernel<<<1, 256, 0, stream>>>(
        cnt_w, cnt_m, ds_w, ds_m, out + 2 * NF, 1.0f / (float)NF);
}
